// LocallyConnected1d_62835371540838
// MI455X (gfx1250) — matrix operand order repaired and checked by reading, not compiled, not run on hardware
//
#include <hip/hip_runtime.h>

// ---------------------------------------------------------------------------
// LocallyConnected1d on MI455X (gfx1250, wave32, WMMA).
// Per output position p: GEMM  out[128x64] = window[128x512] * W[512x64]
// (kernel taps padded 7->8 with zero weight so K' = 512 = 16 x (K=32) steps).
// bf16 inputs via v_wmma_f32_16x16x32_bf16, fp32 accumulate.
// Round 3: sched_group_barrier pipeline hints -- issue all 20 fragment
// ds_load_b128s, then the 8 WMMAs back-to-back (one dscnt wait per chunk).
// ---------------------------------------------------------------------------

typedef __attribute__((ext_vector_type(16))) __bf16 v16bf;
typedef __attribute__((ext_vector_type(8)))  float  v8f;

union Frag { v16bf v; uint4 q[2]; };
static_assert(sizeof(Frag) == 32, "frag must be 8 VGPRs");

constexpr int BATCH = 128;   // GEMM M
constexpr int IC    = 64;    // input channels
constexpr int OC    = 64;    // GEMM N
constexpr int PSZ   = 1024;  // positions
constexpr int KW    = 7;     // real taps
constexpr int TAP8  = 8;     // padded taps (8th weight tap = 0)
constexpr int CHC   = 8;     // channels per pipeline stage
constexpr int CK    = CHC * TAP8;   // 64 k-indices per stage = 2 WMMA K-steps
constexpr int NCH   = IC / CHC;     // 8 stages
constexpr int ASTR  = CK + 8;       // 72 halfs = 144B row: 16B aligned,
constexpr int BSTR  = CK + 8;       //   bank-cycling stride for ds_load_b128

// fp32 -> bf16 (round-half-up) and pack a pair into one dword.
__device__ __forceinline__ unsigned pk_bf16(float a, float b) {
  unsigned ua = __float_as_uint(a) + 0x8000u;
  unsigned ub = __float_as_uint(b) + 0x8000u;
  return (ua >> 16) | (ub & 0xFFFF0000u);
}

__global__ __launch_bounds__(256)
void lc1d_wmma_kernel(const float* __restrict__ x,     // (128, 64, 1024)
                      const float* __restrict__ w,     // (64, 64, 1024, 7)
                      const float* __restrict__ bias,  // (64, 1024)
                      float* __restrict__ out) {       // (128, 64, 1024)
  __shared__ __align__(16) unsigned short Ash[2][BATCH * ASTR]; // 2x18432B
  __shared__ __align__(16) unsigned short Bsh[2][OC    * BSTR]; // 2x 9216B

  const int p    = blockIdx.x;
  const int tid  = threadIdx.x;
  const int wv   = tid >> 5;      // wave id 0..7 -> batch tile
  const int lane = tid & 31;
  const int row  = lane & 15;     // M (A) / N (B,C) row within tile
  const int hi   = lane >> 4;     // lane-half selector in WMMA layouts

  // cooperative-load coordinates: tid -> (row, channel-in-chunk)
  const int c_loc = tid & 7;
  const int r0    = tid >> 3;     // 0..31, +32j

  float ra[4][TAP8];  // staged x windows : 4 batch rows x 8 taps
  float rb[2][KW];    // staged weights   : 2 out-chans  x 7 taps

  v8f acc[4];
  #pragma unroll
  for (int nt = 0; nt < 4; ++nt)
    #pragma unroll
    for (int i = 0; i < 8; ++i) acc[nt][i] = 0.f;

  // ---- stage loaders -------------------------------------------------------
  auto gload = [&](int cc) {              // global -> registers (chunk cc)
    const int c = cc * CHC + c_loc;
    #pragma unroll
    for (int j = 0; j < 4; ++j) {
      const int b = r0 + 32 * j;
      const float* src = x + ((size_t)b * IC + c) * PSZ + (p - 3);
      #pragma unroll
      for (int k = 0; k < TAP8; ++k) {
        const int t = p - 3 + k;          // uniform per block -> scalar branch
        ra[j][k] = ((unsigned)t < (unsigned)PSZ) ? src[k] : 0.f;
      }
    }
    #pragma unroll
    for (int j = 0; j < 2; ++j) {
      const int o = r0 + 32 * j;
      const float* src = w + (((size_t)o * IC + c) * PSZ + p) * KW;
      #pragma unroll
      for (int k = 0; k < KW; ++k) rb[j][k] = src[k];
    }
  };

  auto lstore = [&](int buf) {            // registers -> bf16 LDS tiles
    #pragma unroll
    for (int j = 0; j < 4; ++j) {
      uint4 q;
      q.x = pk_bf16(ra[j][0], ra[j][1]);
      q.y = pk_bf16(ra[j][2], ra[j][3]);
      q.z = pk_bf16(ra[j][4], ra[j][5]);
      q.w = pk_bf16(ra[j][6], ra[j][7]);
      *(uint4*)&Ash[buf][(r0 + 32 * j) * ASTR + c_loc * TAP8] = q;
    }
    #pragma unroll
    for (int j = 0; j < 2; ++j) {
      uint4 q;
      q.x = pk_bf16(rb[j][0], rb[j][1]);
      q.y = pk_bf16(rb[j][2], rb[j][3]);
      q.z = pk_bf16(rb[j][4], rb[j][5]);
      q.w = pk_bf16(rb[j][6], 0.f);       // padded 8th tap kills A's extra elem
      *(uint4*)&Bsh[buf][(r0 + 32 * j) * BSTR + c_loc * TAP8] = q;
    }
  };

  // ---- per-chunk matrix work -----------------------------------------------
  // 2 A frags + 8 B frags (20 ds_load_b128), then 8 WMMAs. sched_group_barrier
  // pins that order so the WMMA chain runs behind a single dscnt wait.
  auto compute = [&](int buf) {
    const unsigned short* As = &Ash[buf][(wv * 16 + row) * ASTR];
    Frag a[2];                            // 16-bit A 16x32 lane layout
    #pragma unroll
    for (int s = 0; s < 2; ++s) {
      a[s].q[0] = *(const uint4*)(As + s * 32 + hi * 16);     // K=16*hi+0..7
      a[s].q[1] = *(const uint4*)(As + s * 32 + hi * 16 + 8); // K=16*hi+8..15
    }
    Frag g[4][2];                         // 16-bit B 32x16: K = 16*hi + 0..15
    #pragma unroll
    for (int nt = 0; nt < 4; ++nt) {
      const unsigned short* Bs = &Bsh[buf][(nt * 16 + row) * BSTR];
      #pragma unroll
      for (int s = 0; s < 2; ++s) {
        g[nt][s].q[0] = *(const uint4*)(Bs + s * 32 + hi * 16);
        g[nt][s].q[1] = *(const uint4*)(Bs + s * 32 + hi * 16 + 8);
      }
    }
    #pragma unroll
    for (int nt = 0; nt < 4; ++nt) {
      acc[nt] = __builtin_amdgcn_wmma_f32_16x16x32_bf16(
          false, a[0].v, false, g[nt][0].v, (short)0, acc[nt], false, false);
      acc[nt] = __builtin_amdgcn_wmma_f32_16x16x32_bf16(
          false, a[1].v, false, g[nt][1].v, (short)0, acc[nt], false, false);
    }
    // Scheduler pipeline: all DS reads first, then the WMMA chain.
    __builtin_amdgcn_sched_group_barrier(0x100, 20, 0); // 20x DS read
    __builtin_amdgcn_sched_group_barrier(0x008,  8, 0); // 8x  WMMA
  };

  // ---- software pipeline: prefetch chunk i+1 while WMMAs chew chunk i ------
  gload(0);
  lstore(0);
  __syncthreads();

  #pragma unroll 2
  for (int ch = 0; ch < NCH; ++ch) {
    if (ch + 1 < NCH) gload(ch + 1);      // VMEM issued under matrix work
    compute(ch & 1);
    if (ch + 1 < NCH) lstore((ch + 1) & 1);
    __syncthreads();
  }

  // ---- epilogue: C/D layout -> out[b,o,p] + bias[o,p] ----------------------
  #pragma unroll
  for (int nt = 0; nt < 4; ++nt) {
    const int o  = nt * 16 + row;
    const float bv = bias[(size_t)o * PSZ + p];
    #pragma unroll
    for (int i = 0; i < 8; ++i) {
      const int b = wv * 16 + hi * 8 + i; // VGPR i: M=i (lanes 0-15) / i+8
      out[((size_t)b * OC + o) * PSZ + p] = acc[nt][i] + bv;
    }
  }
}

extern "C" void kernel_launch(void* const* d_in, const int* in_sizes, int n_in,
                              void* d_out, int out_size, void* d_ws, size_t ws_size,
                              hipStream_t stream) {
  (void)in_sizes; (void)n_in; (void)d_ws; (void)ws_size; (void)out_size;
  const float* x    = (const float*)d_in[0];
  const float* wgt  = (const float*)d_in[1];
  const float* bias = (const float*)d_in[2];
  float* out        = (float*)d_out;
  lc1d_wmma_kernel<<<dim3(PSZ), dim3(256), 0, stream>>>(x, wgt, bias, out);
}
